// LatticeGCN_46772193853800
// MI455X (gfx1250) — matrix operand order repaired and checked by reading, not compiled, not run on hardware
//
#include <hip/hip_runtime.h>
#include <stdint.h>

#define DIN 256
#define DH  128
#define DE  64

typedef __attribute__((ext_vector_type(16))) __bf16       v16bf;
typedef __attribute__((ext_vector_type(8)))  float        v8f;
typedef __attribute__((ext_vector_type(4)))  unsigned int u32x4;
typedef __attribute__((ext_vector_type(4)))  float        f32x4;

__device__ __forceinline__ unsigned short f32_to_bf16(float f) {
  unsigned int u = __float_as_uint(f);
  u += 0x7FFFu + ((u >> 16) & 1u);   // round-to-nearest-even
  return (unsigned short)(u >> 16);
}

union FragAB { u32x4 u[2]; v16bf v; };

// ---------------------------------------------------------------------------
// C[M,Nout] = A[M,K](bf16) * Bt[Nout,K](bf16)^T  (+bias)(+relu)
// One wave = one 16x16 output tile, v_wmma_f32_16x16x32_bf16 over K.
// A-frag layout (16-bit A 16x32): lane m=lane&15, hi=lane>>4:
//   elems 0..7 -> K = k0 + hi*8 + i ; elems 8..15 -> K = k0 + 16 + hi*8 + i
// B-frag layout (16-bit B 32x16): lane n=lane&15, hi=lane>>4 (same as A):
//   elems 0..7 -> K = k0 + hi*8 + i ; elems 8..15 -> K = k0 + 16 + hi*8 + i
// ---------------------------------------------------------------------------
__global__ __launch_bounds__(128)
void k_gemm_bf16(const unsigned short* __restrict__ A,
                 const unsigned short* __restrict__ Bt,
                 const float* __restrict__ bias,
                 float* __restrict__ Cf,
                 unsigned short* __restrict__ Cb,
                 int M, int K, int Nout, int relu)
{
  const int wave    = threadIdx.x >> 5;
  const int lane    = threadIdx.x & 31;
  const int rowTile = blockIdx.x * 4 + wave;
  if (rowTile * 16 >= M) return;            // wave-uniform: EXEC stays all-1s
  const int colTile = blockIdx.y;
  const int mn = lane & 15;
  const int hi = lane >> 4;

  const unsigned short* arow = A  + (size_t)(rowTile * 16 + mn) * K;
  const unsigned short* brow = Bt + (size_t)(colTile * 16 + mn) * K;

  v8f acc = {0.f, 0.f, 0.f, 0.f, 0.f, 0.f, 0.f, 0.f};

  for (int k = 0; k < K; k += 32) {
    FragAB a, b;
    a.u[0] = *(const u32x4*)(arow + k + hi * 8);
    a.u[1] = *(const u32x4*)(arow + k + 16 + hi * 8);
    b.u[0] = *(const u32x4*)(brow + k + hi * 8);
    b.u[1] = *(const u32x4*)(brow + k + 16 + hi * 8);
    acc = __builtin_amdgcn_wmma_f32_16x16x32_bf16(false, a.v, false, b.v,
                                                  (short)0, acc, false, false);
  }

  // D layout: lane holds column n=lane&15; VGPR r holds row r + hi*8
  const int cg = colTile * 16 + mn;
  const float bv = bias ? bias[cg] : 0.0f;
#pragma unroll
  for (int r = 0; r < 8; ++r) {
    const int mg = rowTile * 16 + hi * 8 + r;
    float v = acc[r] + bv;
    if (relu) v = fmaxf(v, 0.0f);
    const size_t off = (size_t)mg * Nout + cg;
    if (Cf) Cf[off] = v;
    if (Cb) Cb[off] = f32_to_bf16(v);
  }
}

// --------------------------- small utility kernels -------------------------
__global__ void k_fill1(float* a, int n) {
  int i = blockIdx.x * blockDim.x + threadIdx.x;
  if (i < n) a[i] = 1.0f;
}
__global__ void k_deg(const int* __restrict__ dst, float* deg, int E) {
  int e = blockIdx.x * blockDim.x + threadIdx.x;
  if (e < E) unsafeAtomicAdd(&deg[dst[e]], 1.0f);
}
__global__ void k_rsqrt_inplace(float* a, int n) {
  int i = blockIdx.x * blockDim.x + threadIdx.x;
  if (i < n) a[i] = rsqrtf(a[i]);
}
__global__ void k_coef(const int* __restrict__ s, const int* __restrict__ t,
                       const float* __restrict__ dinv, float* coef, int E) {
  int e = blockIdx.x * blockDim.x + threadIdx.x;
  if (e < E) coef[e] = dinv[s[e]] * dinv[t[e]];
}
__global__ void k_cvt_bf16(const float* __restrict__ in, unsigned short* out, long long n) {
  long long i = (long long)blockIdx.x * blockDim.x + threadIdx.x;
  if (i < n) out[i] = f32_to_bf16(in[i]);
}
// Wt[n*K+k] = bf16(W[k*Nout+n])
__global__ void k_wt(const float* __restrict__ W, unsigned short* Wt, int K, int Nout) {
  int i = blockIdx.x * blockDim.x + threadIdx.x;
  if (i < K * Nout) {
    int k = i / Nout, n = i % Nout;
    Wt[(size_t)n * K + k] = f32_to_bf16(W[i]);
  }
}
// agg = h * dinv^2 + b   (self-loop term + bias, also initializes agg)
__global__ void k_self_bias(const float* __restrict__ h, const float* __restrict__ dinv,
                            const float* __restrict__ b, float* agg, int N, int D) {
  long long i = (long long)blockIdx.x * blockDim.x + threadIdx.x;
  if (i < (long long)N * D) {
    int n = (int)(i / D), d = (int)(i % D);
    float di = dinv[n];
    agg[i] = h[i] * di * di + b[d];
  }
}
// agg[dst] += h[src] * coef  (float4 gather + 4 hw FP atomics)
__global__ void k_scatter(const float* __restrict__ h, const int* __restrict__ src,
                          const int* __restrict__ dst, const float* __restrict__ coef,
                          float* agg, int E, int D) {
  int e = blockIdx.x * blockDim.x + threadIdx.x;
  if (e >= E) return;
  int c = blockIdx.y * 4;
  int s = src[e], t = dst[e];
  float cf = coef[e];
  f32x4 hv = *(const f32x4*)(h + (size_t)s * D + c);
  float* out = agg + (size_t)t * D + c;
  unsafeAtomicAdd(out + 0, hv[0] * cf);
  unsafeAtomicAdd(out + 1, hv[1] * cf);
  unsafeAtomicAdd(out + 2, hv[2] * cf);
  unsafeAtomicAdd(out + 3, hv[3] * cf);
}
__global__ void k_relu_cvt(const float* __restrict__ a, unsigned short* o, long long n) {
  long long i = (long long)blockIdx.x * blockDim.x + threadIdx.x;
  if (i < n) o[i] = f32_to_bf16(fmaxf(a[i], 0.0f));
}
// z: f32 to d_out and bf16 copy for the decoder (no relu)
__global__ void k_z_out(const float* __restrict__ a, float* zf, unsigned short* zb, long long n) {
  long long i = (long long)blockIdx.x * blockDim.x + threadIdx.x;
  if (i < n) { float v = a[i]; zf[i] = v; zb[i] = f32_to_bf16(v); }
}

// ---------------------------------------------------------------------------
extern "C" void kernel_launch(void* const* d_in, const int* in_sizes, int n_in,
                              void* d_out, int out_size, void* d_ws, size_t ws_size,
                              hipStream_t stream) {
  const float* x   = (const float*)d_in[0];
  const int*   ei  = (const int*)d_in[1];
  const float* W1  = (const float*)d_in[2];
  const float* b1  = (const float*)d_in[3];
  const float* W2  = (const float*)d_in[4];
  const float* b2  = (const float*)d_in[5];
  const float* W3  = (const float*)d_in[6];
  const float* b3  = (const float*)d_in[7];
  const float* Wd1 = (const float*)d_in[8];
  const float* bd1 = (const float*)d_in[9];
  const float* Wd2 = (const float*)d_in[10];
  const float* bd2 = (const float*)d_in[11];

  const int N = in_sizes[0] / DIN;
  const int E = in_sizes[1] / 2;
  const int* src = ei;
  const int* dst = ei + E;

  // ---- carve workspace (256B-aligned regions) ----
  char* p = (char*)d_ws;
  auto carve = [&](size_t bytes) -> void* {
    void* r = (void*)p;
    p += (bytes + 255) & ~(size_t)255;
    return r;
  };
  unsigned short* actsA = (unsigned short*)carve((size_t)N * DIN * 2); // ping (<=256 wide)
  unsigned short* actsB = (unsigned short*)carve((size_t)N * DH  * 2); // pong (<=128 wide)
  float* hbuf = (float*)carve((size_t)N * DH * 4);
  float* agg  = (float*)carve((size_t)N * DH * 4);
  float* dinv = (float*)carve((size_t)N * 4);   // deg -> rsqrt in place
  float* coef = (float*)carve((size_t)E * 4);
  unsigned short* W1t  = (unsigned short*)carve((size_t)DIN * DH * 2);
  unsigned short* W2t  = (unsigned short*)carve((size_t)DH  * DH * 2);
  unsigned short* W3t  = (unsigned short*)carve((size_t)DH  * DE * 2);
  unsigned short* Wd1t = (unsigned short*)carve((size_t)DE  * DH * 2);
  unsigned short* Wd2t = (unsigned short*)carve((size_t)DH  * DIN * 2);

  const dim3 blk(256);
  auto g1 = [](long long t) { return dim3((unsigned)((t + 255) / 256)); };

  // ---- degrees / norm coefficients ----
  k_fill1<<<g1(N), blk, 0, stream>>>(dinv, N);
  k_deg<<<g1(E), blk, 0, stream>>>(dst, dinv, E);
  k_rsqrt_inplace<<<g1(N), blk, 0, stream>>>(dinv, N);
  k_coef<<<g1(E), blk, 0, stream>>>(src, dst, dinv, coef, E);

  // ---- bf16 conversions (input + transposed weights) ----
  k_cvt_bf16<<<g1((long long)N * DIN), blk, 0, stream>>>(x, actsA, (long long)N * DIN);
  k_wt<<<g1(DIN * DH), blk, 0, stream>>>(W1,  W1t,  DIN, DH);
  k_wt<<<g1(DH  * DH), blk, 0, stream>>>(W2,  W2t,  DH,  DH);
  k_wt<<<g1(DH  * DE), blk, 0, stream>>>(W3,  W3t,  DH,  DE);
  k_wt<<<g1(DE  * DH), blk, 0, stream>>>(Wd1, Wd1t, DE,  DH);
  k_wt<<<g1(DH  * DIN), blk, 0, stream>>>(Wd2, Wd2t, DH,  DIN);

  const unsigned rowBlks = (unsigned)((N / 16 + 3) / 4);
  const dim3 gblk(128);

  // ---- GCN layer 1: [N,256] -> [N,128], relu ----
  k_gemm_bf16<<<dim3(rowBlks, DH / 16), gblk, 0, stream>>>(actsA, W1t, nullptr, hbuf, nullptr, N, DIN, DH, 0);
  k_self_bias<<<g1((long long)N * DH), blk, 0, stream>>>(hbuf, dinv, b1, agg, N, DH);
  k_scatter<<<dim3((unsigned)((E + 255) / 256), DH / 4), blk, 0, stream>>>(hbuf, src, dst, coef, agg, E, DH);
  k_relu_cvt<<<g1((long long)N * DH), blk, 0, stream>>>(agg, actsB, (long long)N * DH);

  // ---- GCN layer 2: [N,128] -> [N,128], relu ----
  k_gemm_bf16<<<dim3(rowBlks, DH / 16), gblk, 0, stream>>>(actsB, W2t, nullptr, hbuf, nullptr, N, DH, DH, 0);
  k_self_bias<<<g1((long long)N * DH), blk, 0, stream>>>(hbuf, dinv, b2, agg, N, DH);
  k_scatter<<<dim3((unsigned)((E + 255) / 256), DH / 4), blk, 0, stream>>>(hbuf, src, dst, coef, agg, E, DH);
  k_relu_cvt<<<g1((long long)N * DH), blk, 0, stream>>>(agg, actsA, (long long)N * DH);

  // ---- GCN layer 3: [N,128] -> z [N,64], no relu ----
  k_gemm_bf16<<<dim3(rowBlks, DE / 16), gblk, 0, stream>>>(actsA, W3t, nullptr, hbuf, nullptr, N, DH, DE, 0);
  k_self_bias<<<g1((long long)N * DE), blk, 0, stream>>>(hbuf, dinv, b3, agg, N, DE);
  k_scatter<<<dim3((unsigned)((E + 255) / 256), DE / 4), blk, 0, stream>>>(hbuf, src, dst, coef, agg, E, DE);
  float* zout = (float*)d_out;                               // first output: z [N,64]
  k_z_out<<<g1((long long)N * DE), blk, 0, stream>>>(agg, zout, actsB, (long long)N * DE);

  // ---- decoder: relu(z@Wd1+bd1) @ Wd2 + bd2 ----
  k_gemm_bf16<<<dim3(rowBlks, DH / 16), gblk, 0, stream>>>(actsB, Wd1t, bd1, nullptr, actsA, N, DE, DH, 1);
  float* xhat = (float*)d_out + (size_t)N * DE;              // second output: x_hat [N,256]
  k_gemm_bf16<<<dim3(rowBlks, DIN / 16), gblk, 0, stream>>>(actsA, Wd2t, bd2, xhat, nullptr, N, DH, DIN, 0);
}
